// MultiHeadAttention_50964081934633
// MI455X (gfx1250) — matrix operand order repaired and checked by reading, not compiled, not run on hardware
//
#include <hip/hip_runtime.h>
#include <hip/hip_bf16.h>

// ---------------------------------------------------------------------------
// MI455X (gfx1250) multi-head attention, wave32 + v_wmma_f32_16x16x32_f16.
// Pipeline: 3x projection GEMM (f32 -> f16 WMMA) -> flash attention -> output
// projection GEMM. All matrix contractions (including softmax row-sums) run
// on the matrix cores. Attention K/V blocks are staged to LDS once per
// workgroup with gfx1250 async global->LDS loads (ASYNCcnt) and shared by
// all 8 waves.
// ---------------------------------------------------------------------------

typedef _Float16 h16;
typedef __attribute__((ext_vector_type(16))) _Float16 v16h;
typedef __attribute__((ext_vector_type(8)))  float    v8f;

union Frag16 { v16h h; uint4 q[2]; };

#define S_LEN   2048
#define DMODEL  1024
#define NHEAD   16
#define DK      64
#define M_TOT   4096   // B * S

__device__ __forceinline__ unsigned pack2f16(float a, float b) {
    union { h16 h[2]; unsigned u; } p;
    p.h[0] = (h16)a; p.h[1] = (h16)b;
    return p.u;
}

__device__ __forceinline__ v8f wmma_f16(const Frag16& a, const Frag16& b, v8f c) {
    // D = A(16x32 f16) * B(32x16 f16) + C(16x16 f32)
    return __builtin_amdgcn_wmma_f32_16x16x32_f16(
        /*neg_a=*/false, a.h, /*neg_b=*/false, b.h,
        /*c_mod=*/(short)0, c, /*reuse_a=*/false, /*reuse_b=*/false);
}

// gfx1250 async global->LDS copy, 16B per active lane; tracked by ASYNCcnt.
__device__ __forceinline__ void async_ld_b128(unsigned lds_off, const void* g) {
    asm volatile("global_load_async_to_lds_b128 %0, %1, off"
                 :: "v"(lds_off), "v"(g) : "memory");
}

// ---------------------------------------------------------------------------
// GEMM: Y[m][n] = sum_k A[m][k] * W[n][k]   (i.e. Y = A @ W^T), M=4096,K=N=1024
// MODE 0: A f32, write f16 head-split   out[(b*16+h)][s][d]
// MODE 1: A f32, write f16 transposed   out[(b*16+h)][d][s]      (for V)
// MODE 2: A f16, write f32 plain        out[m][n]                (final proj)
// Block = 256 thr (8 waves), tile 128x128, K-step 32; wave tile 32x64.
// ---------------------------------------------------------------------------
template<int MODE>
__global__ __launch_bounds__(256) void gemm_wmma(
    const void* __restrict__ Ap, const float* __restrict__ W, void* __restrict__ outp)
{
    __shared__ alignas(16) unsigned char ldsA[128 * 64]; // 128 rows x 32 f16
    __shared__ alignas(16) unsigned char ldsB[128 * 64];

    const int tid  = threadIdx.x;
    const int wid  = tid >> 5;
    const int lane = tid & 31;
    const int lrow = lane & 15;
    const int hlf  = lane >> 4;
    const int m0 = blockIdx.x * 128;
    const int n0 = blockIdx.y * 128;
    const int wm = (wid >> 1) * 32;   // wave m-offset in tile
    const int wn = (wid & 1) * 64;    // wave n-offset in tile

    const v8f vzero = {};
    v8f acc[2][4];
#pragma unroll
    for (int i = 0; i < 2; ++i)
#pragma unroll
        for (int j = 0; j < 4; ++j) acc[i][j] = vzero;

    for (int k0 = 0; k0 < DMODEL; k0 += 32) {
        // ---- stage A tile (convert f32 -> f16 on the fly for MODE<2) ----
        if constexpr (MODE < 2) {
            const float* A = (const float*)Ap;
#pragma unroll
            for (int i = 0; i < 4; ++i) {
                const int f = tid + i * 256;          // float4 id 0..1023
                const int row = f >> 3, cq = f & 7;
                const float4 v = *(const float4*)(A + (size_t)(m0 + row) * DMODEL + k0 + cq * 4);
                unsigned* d = (unsigned*)(ldsA + row * 64 + cq * 8);
                d[0] = pack2f16(v.x, v.y);
                d[1] = pack2f16(v.z, v.w);
            }
        } else {
            const h16* A = (const h16*)Ap;
#pragma unroll
            for (int i = 0; i < 2; ++i) {
                const int f = tid + i * 256;          // uint4 id 0..511
                const int row = f >> 2, cq = f & 3;
                *(uint4*)(ldsA + row * 64 + cq * 16) =
                    *(const uint4*)(A + (size_t)(m0 + row) * DMODEL + k0 + cq * 8);
            }
        }
        // ---- stage W tile (rows = output features n0..n0+127) ----
#pragma unroll
        for (int i = 0; i < 4; ++i) {
            const int f = tid + i * 256;
            const int row = f >> 3, cq = f & 7;
            const float4 v = *(const float4*)(W + (size_t)(n0 + row) * DMODEL + k0 + cq * 4);
            unsigned* d = (unsigned*)(ldsB + row * 64 + cq * 8);
            d[0] = pack2f16(v.x, v.y);
            d[1] = pack2f16(v.z, v.w);
        }
        if (k0 + 32 < DMODEL) {
            if constexpr (MODE < 2)
                __builtin_prefetch((const float*)Ap + (size_t)(m0 + (tid >> 3)) * DMODEL + k0 + 32, 0, 1);
            __builtin_prefetch(W + (size_t)(n0 + (tid >> 3)) * DMODEL + k0 + 32, 0, 1);
        }
        __syncthreads();

        // ---- fragments from LDS (ISA 16-bit A 16x32 / B 32x16 layouts) ----
        Frag16 a[2], b[4];
#pragma unroll
        for (int i = 0; i < 2; ++i) {
            const unsigned char* base = ldsA + (wm + i * 16 + lrow) * 64;
            a[i].q[0] = *(const uint4*)(base + hlf * 16);        // K = 8h..8h+7
            a[i].q[1] = *(const uint4*)(base + 32 + hlf * 16);   // K = 16+8h..
        }
#pragma unroll
        for (int j = 0; j < 4; ++j) {
            const unsigned char* base = ldsB + (wn + j * 16 + lrow) * 64;
            b[j].q[0] = *(const uint4*)(base + hlf * 16);        // K = 8h..8h+7
            b[j].q[1] = *(const uint4*)(base + 32 + hlf * 16);   // K = 16+8h..
        }
#pragma unroll
        for (int i = 0; i < 2; ++i)
#pragma unroll
            for (int j = 0; j < 4; ++j)
                acc[i][j] = wmma_f16(a[i], b[j], acc[i][j]);
        __syncthreads();
    }

    // ---- epilogue: C/D layout lane holds rows r + 8*hlf, col lane%16 ----
#pragma unroll
    for (int i = 0; i < 2; ++i) {
#pragma unroll
        for (int j = 0; j < 4; ++j) {
            const int gmb = m0 + wm + i * 16 + hlf * 8;
            const int gn  = n0 + wn + j * 16 + lrow;
#pragma unroll
            for (int r = 0; r < 8; ++r) {
                const int gm = gmb + r;
                const float val = acc[i][j][r];
                if constexpr (MODE == 2) {
                    ((float*)outp)[(size_t)gm * DMODEL + gn] = val;
                } else {
                    const int bb = gm >> 11, s = gm & 2047;   // m = b*2048+s
                    const int hh = gn >> 6,  d = gn & 63;     // n = h*64+d
                    h16* o = (h16*)outp;
                    if constexpr (MODE == 0)
                        o[(((size_t)(bb * NHEAD + hh) * S_LEN + s) << 6) + d] = (h16)val;
                    else
                        o[((size_t)(bb * NHEAD + hh) * DK + d) * S_LEN + s] = (h16)val;
                }
            }
        }
    }
}

// ---------------------------------------------------------------------------
// Flash attention per (b,h, 128-row q-block). 8 waves, 16 q-rows each,
// 64 keys per iteration:
//   - K (64x64 f16) and V^T (64x64 f16) staged to LDS ONCE per workgroup via
//     gfx1250 async global->LDS b128 copies (ASYNCcnt), shared by all 8 waves
//   - 8 WMMA for S = Q K^T (Q pre-scaled by 1/sqrt(dk), exact pow2 in f16)
//   - row-max via 4 shuffles/row (only remaining cross-lane reduction)
//   - P -> LDS -> A-fragment relayout
//   - 2 WMMA vs a ones-matrix produce softmax row-sums (no shuffle reduction)
//   - 8 WMMA for O += P V
// LDS rows padded to 144B (9 x 16B banks-groups) to kill b128 read conflicts.
// Context written f16 [B,S,1024].
// ---------------------------------------------------------------------------
#define KVSTR 144          // padded byte stride for 64 f16 (128B) rows

__global__ __launch_bounds__(256) void attn_wmma(
    const h16* __restrict__ Qh, const h16* __restrict__ Kh, const h16* __restrict__ Vt,
    const int* __restrict__ mask, h16* __restrict__ Ctx)
{
    __shared__ alignas(16) unsigned char ldsK[64 * KVSTR];   // [key][d]
    __shared__ alignas(16) unsigned char ldsV[64 * KVSTR];   // [d][key]
    __shared__ alignas(16) unsigned char ldsP[8 * 16 * KVSTR]; // per-wave 16x64

    const int tid  = threadIdx.x;
    const int wid  = tid >> 5;
    const int lane = tid & 31;
    const int lrow = lane & 15;
    const int hlf  = lane >> 4;
    const int bh = blockIdx.x;            // b*16 + h
    const int qb = blockIdx.y;
    const int q0 = qb * 128 + wid * 16;
    const int bb = bh >> 4;
    const int hh = bh & 15;
    const h16* Qp = Qh + (size_t)bh * S_LEN * DK;
    const h16* Kp = Kh + (size_t)bh * S_LEN * DK;
    const h16* Vp = Vt + (size_t)bh * DK * S_LEN;  // [d][s]
    const int* mrow = mask + bb * S_LEN;
    unsigned char* myP = ldsP + wid * 16 * KVSTR;

    // LDS byte offsets (low 32 bits of generic address == LDS offset)
    const unsigned kBase = (unsigned)(uintptr_t)(void*)ldsK;
    const unsigned vBase = (unsigned)(uintptr_t)(void*)ldsV;
    // each thread stages 2 x 16B of K and 2 x 16B of V per 64-key block
    const int crow0 = tid >> 3, ccol = tid & 7;          // rows 0..31
    const int crow1 = crow0 + 32;                        // rows 32..63

    // Q A-fragments (d = 0..31 and 32..63), pre-scaled by 1/sqrt(64) = 0.125
    Frag16 aq0, aq1;
    {
        const h16* qrow = Qp + (size_t)(q0 + lrow) * DK;
        aq0.q[0] = *(const uint4*)(qrow + hlf * 8);
        aq0.q[1] = *(const uint4*)(qrow + 16 + hlf * 8);
        aq1.q[0] = *(const uint4*)(qrow + 32 + hlf * 8);
        aq1.q[1] = *(const uint4*)(qrow + 48 + hlf * 8);
#pragma unroll
        for (int i = 0; i < 16; ++i) {
            aq0.h[i] *= (h16)0.125f;
            aq1.h[i] *= (h16)0.125f;
        }
    }

    // ones B-matrix for row-sum WMMA (f16 1.0 = 0x3C00)
    Frag16 onesB;
    onesB.q[0] = make_uint4(0x3C003C00u, 0x3C003C00u, 0x3C003C00u, 0x3C003C00u);
    onesB.q[1] = onesB.q[0];

    const v8f vzero = {};
    v8f o[4], lacc;
#pragma unroll
    for (int t = 0; t < 4; ++t) o[t] = vzero;
    lacc = vzero;
    float mrun[8];
#pragma unroll
    for (int r = 0; r < 8; ++r) mrun[r] = -1e30f;

    for (int kb = 0; kb < S_LEN; kb += 64) {
        // ---- async stage K[kb..kb+63][0..63] and Vt[0..63][kb..kb+63] ----
        async_ld_b128(kBase + crow0 * KVSTR + ccol * 16,
                      Kp + (size_t)(kb + crow0) * DK + ccol * 8);
        async_ld_b128(kBase + crow1 * KVSTR + ccol * 16,
                      Kp + (size_t)(kb + crow1) * DK + ccol * 8);
        async_ld_b128(vBase + crow0 * KVSTR + ccol * 16,
                      Vp + (size_t)crow0 * S_LEN + kb + ccol * 8);
        async_ld_b128(vBase + crow1 * KVSTR + ccol * 16,
                      Vp + (size_t)crow1 * S_LEN + kb + ccol * 8);
        asm volatile("s_wait_asynccnt 0" ::: "memory");
        __syncthreads();

        // ---- S = Q K^T over 4 key sub-blocks of 16 (K frags from LDS) ----
        v8f s[4];
#pragma unroll
        for (int t = 0; t < 4; ++t) {
            const unsigned char* krow = ldsK + (t * 16 + lrow) * KVSTR;
            Frag16 b0, b1;
            b0.q[0] = *(const uint4*)(krow + hlf * 16);        // d = 8h..
            b0.q[1] = *(const uint4*)(krow + 32 + hlf * 16);
            b1.q[0] = *(const uint4*)(krow + 64 + hlf * 16);   // d = 32+8h..
            b1.q[1] = *(const uint4*)(krow + 96 + hlf * 16);
            s[t] = wmma_f16(aq0, b0, vzero);
            s[t] = wmma_f16(aq1, b1, s[t]);
        }
        // ---- mask (scores already scaled; reference masks after scaling) ----
#pragma unroll
        for (int t = 0; t < 4; ++t) {
            const bool mz = (mrow[kb + t * 16 + lrow] == 0);
#pragma unroll
            for (int r = 0; r < 8; ++r)
                s[t][r] = mz ? -1e9f : s[t][r];
        }
        // ---- online row-max (4 shuffles per row) + alpha ----
        float alpha[8];
#pragma unroll
        for (int r = 0; r < 8; ++r) {
            float mx = fmaxf(fmaxf(s[0][r], s[1][r]), fmaxf(s[2][r], s[3][r]));
            mx = fmaxf(mx, __shfl_xor(mx, 8, 16));
            mx = fmaxf(mx, __shfl_xor(mx, 4, 16));
            mx = fmaxf(mx, __shfl_xor(mx, 2, 16));
            mx = fmaxf(mx, __shfl_xor(mx, 1, 16));
            const float mnew = fmaxf(mrun[r], mx);
            alpha[r] = __expf(mrun[r] - mnew);
            mrun[r] = mnew;
        }
        // ---- P = exp(S - m) -> LDS (16 rows x 64 keys, KVSTR row stride) ----
#pragma unroll
        for (int r = 0; r < 8; ++r) {
            h16* pp = (h16*)(myP + (r + 8 * hlf) * KVSTR) + lrow;
#pragma unroll
            for (int t = 0; t < 4; ++t)
                pp[t * 16] = (h16)__expf(s[t][r] - mrun[r]);
        }
        // ---- rescale running accumulators ----
#pragma unroll
        for (int r = 0; r < 8; ++r) {
            lacc[r] *= alpha[r];
#pragma unroll
            for (int t = 0; t < 4; ++t) o[t][r] *= alpha[r];
        }
        // cross-lane LDS store->load inside the wave: drain DScnt explicitly
        asm volatile("s_wait_dscnt 0" ::: "memory");
        Frag16 ap0, ap1;
        const unsigned char* pb = myP + lrow * KVSTR;
        ap0.q[0] = *(const uint4*)(pb + hlf * 16);        // keys 0..31
        ap0.q[1] = *(const uint4*)(pb + 32 + hlf * 16);
        ap1.q[0] = *(const uint4*)(pb + 64 + hlf * 16);   // keys 32..63
        ap1.q[1] = *(const uint4*)(pb + 96 + hlf * 16);

        // ---- softmax denominators: l += P * ones (row-sum in every column) --
        lacc = wmma_f16(ap0, onesB, lacc);
        lacc = wmma_f16(ap1, onesB, lacc);

        // ---- O += P V, V frags from LDS (ldsV is [d][key]) ----
#pragma unroll
        for (int t = 0; t < 4; ++t) {
            const unsigned char* vrow = ldsV + (t * 16 + lrow) * KVSTR;
            Frag16 bv0, bv1;
            bv0.q[0] = *(const uint4*)(vrow + hlf * 16);        // keys 8h..
            bv0.q[1] = *(const uint4*)(vrow + 32 + hlf * 16);
            bv1.q[0] = *(const uint4*)(vrow + 64 + hlf * 16);   // keys 32+8h..
            bv1.q[1] = *(const uint4*)(vrow + 96 + hlf * 16);
            o[t] = wmma_f16(ap0, bv0, o[t]);
            o[t] = wmma_f16(ap1, bv1, o[t]);
        }
        __syncthreads();   // protect ldsK/ldsV before next stage overwrites
    }

    // normalize and write context f16 as [B, S, H*DK]
#pragma unroll
    for (int r = 0; r < 8; ++r) {
        const float inv = 1.0f / lacc[r];
        const int q = q0 + r + 8 * hlf;
#pragma unroll
        for (int t = 0; t < 4; ++t) {
            const int d = t * 16 + lrow;
            Ctx[((size_t)bb * S_LEN + q) * DMODEL + hh * DK + d] = (h16)(o[t][r] * inv);
        }
    }
}

// ---------------------------------------------------------------------------
extern "C" void kernel_launch(void* const* d_in, const int* in_sizes, int n_in,
                              void* d_out, int out_size, void* d_ws, size_t ws_size,
                              hipStream_t stream) {
    const float* q    = (const float*)d_in[0];
    const float* k    = (const float*)d_in[1];
    const float* v    = (const float*)d_in[2];
    const int*   mask = (const int*)d_in[3];
    const float* wq   = (const float*)d_in[4];
    const float* wk   = (const float*)d_in[5];
    const float* wv   = (const float*)d_in[6];
    const float* wo   = (const float*)d_in[7];

    unsigned char* ws = (unsigned char*)d_ws;
    const size_t HALF_MAT = (size_t)M_TOT * DMODEL * sizeof(h16); // 8 MB each
    h16* Qh  = (h16*)(ws);
    h16* Kh  = (h16*)(ws + HALF_MAT);
    h16* Vt  = (h16*)(ws + 2 * HALF_MAT);
    h16* Ctx = (h16*)(ws + 3 * HALF_MAT);

    const dim3 blk(256);
    const dim3 ggrid(M_TOT / 128, DMODEL / 128);   // 32 x 8 tiles
    gemm_wmma<0><<<ggrid, blk, 0, stream>>>(q, wq, Qh);
    gemm_wmma<0><<<ggrid, blk, 0, stream>>>(k, wk, Kh);
    gemm_wmma<1><<<ggrid, blk, 0, stream>>>(v, wv, Vt);
    attn_wmma<<<dim3(2 * NHEAD, S_LEN / 128), blk, 0, stream>>>(Qh, Kh, Vt, mask, Ctx);
    gemm_wmma<2><<<ggrid, blk, 0, stream>>>(Ctx, wo, d_out);
}
